// Linear_binary_27968827032034
// MI455X (gfx1250) — matrix operand order repaired and checked by reading, not compiled, not run on hardware
//
#include <hip/hip_runtime.h>

// ---------------------------------------------------------------------------
// Linear_binary on MI455X (gfx1250):
//   Phase 1a: greedy binary-basis decomposition of weight -> bf16 w_approx (ws)
//   Phase 1b: x (f32) -> bf16 once (convert-once / read-many)           (ws)
//   Phase 2 : y = x @ w_approx^T + bias, v_wmma_f32_16x16x32_bf16,
//             128x128x64 tiles, 8 wave32/block, double-buffered LDS,
//             BOTH operands streamed with global_load_async_to_lds_b128
//             (ASYNCcnt) -> inner loop is ds_load_b128 + WMMA only.
// ---------------------------------------------------------------------------

#define IN_DIM  4096
#define OUT_DIM 4096
#define NROWS   8192

#define BM  128
#define BN  128
#define BK  64
#define LDK 72   // padded LDS row stride in elements (144 B = multiple of 16 B)
#define BUFB (BM * LDK * 2)   // byte stride between double buffers (18432)

typedef __attribute__((ext_vector_type(16))) __bf16 v16bf;
typedef __attribute__((ext_vector_type(8)))  __bf16 v8bf;
typedef __attribute__((ext_vector_type(4)))  __bf16 bf16x4;
typedef __attribute__((ext_vector_type(8)))  float  v8f;
typedef __attribute__((ext_vector_type(4)))  float  f32x4;

__device__ __forceinline__ unsigned short f2bf(float f) {
  __bf16 h = (__bf16)f;                       // native RTNE convert
  return __builtin_bit_cast(unsigned short, h);
}

// Sum of one value per thread across a 256-thread block (8 wave32).
__device__ __forceinline__ float block_sum(float v, float* sbuf) {
  #pragma unroll
  for (int off = 16; off > 0; off >>= 1) v += __shfl_down(v, off, 32);
  const int lane = threadIdx.x & 31;
  const int wid  = threadIdx.x >> 5;
  if (lane == 0) sbuf[wid] = v;
  __syncthreads();
  if (wid == 0) {
    float t = (lane < 8) ? sbuf[lane] : 0.0f;
    #pragma unroll
    for (int off = 4; off > 0; off >>= 1) t += __shfl_down(t, off, 32);
    if (lane == 0) sbuf[0] = t;
  }
  __syncthreads();
  const float r = sbuf[0];
  __syncthreads();                             // sbuf reused across calls
  return r;
}

// ---------------- Phase 1a: one block per weight row ------------------------
__global__ __launch_bounds__(256) void binary_decompose_kernel(
    const float* __restrict__ w, const int* __restrict__ basis,
    unsigned short* __restrict__ wb) {
  __shared__ float sbuf[8];
  const int row = blockIdx.x;
  const int t   = threadIdx.x;
  const float* wr = w + (size_t)row * IN_DIM;

  float r[16], a[16];
  float s = 0.0f;
  #pragma unroll
  for (int j = 0; j < 16; ++j) { r[j] = wr[j * 256 + t]; s += r[j]; }
  const float offset = block_sum(s, sbuf) * (1.0f / IN_DIM);
  #pragma unroll
  for (int j = 0; j < 16; ++j) { r[j] -= offset; a[j] = offset; }

  const int nb = basis[0];
  for (int k = 0; k < nb; ++k) {
    float sa = 0.0f;
    #pragma unroll
    for (int j = 0; j < 16; ++j) sa += fabsf(r[j]);
    const float c = block_sum(sa, sbuf) * (1.0f / IN_DIM);
    #pragma unroll
    for (int j = 0; j < 16; ++j) {
      const float cb = (r[j] >= 0.0f) ? c : -c;   // c * sign(r)
      a[j] += cb;
      r[j] -= cb;
    }
  }
  unsigned short* o = wb + (size_t)row * IN_DIM;
  #pragma unroll
  for (int j = 0; j < 16; ++j) o[j * 256 + t] = f2bf(a[j]);
}

// ---------------- Phase 1b: x f32 -> bf16, 8 elems/thread -------------------
__global__ __launch_bounds__(256) void cvt_x_bf16_kernel(
    const float* __restrict__ x, unsigned short* __restrict__ xb) {
  const size_t i = ((size_t)blockIdx.x * 256 + threadIdx.x) * 8;
  const f32x4 a = *(const f32x4*)(x + i);
  const f32x4 b = *(const f32x4*)(x + i + 4);
  v8bf c;
  c[0] = (__bf16)a[0]; c[1] = (__bf16)a[1];
  c[2] = (__bf16)a[2]; c[3] = (__bf16)a[3];
  c[4] = (__bf16)b[0]; c[5] = (__bf16)b[1];
  c[6] = (__bf16)b[2]; c[7] = (__bf16)b[3];
  *(v8bf*)(xb + i) = c;
}

// ---------------- Phase 2 (primary): bf16 A + bf16 B, fully async -----------
__global__ __launch_bounds__(256) void gemm_wmma_async_kernel(
    const unsigned short* __restrict__ xb, const unsigned short* __restrict__ wb,
    const float* __restrict__ bias, float* __restrict__ out) {
  __shared__ __align__(16) unsigned short sA[2][BM * LDK];
  __shared__ __align__(16) unsigned short sB[2][BN * LDK];

  const int t    = threadIdx.x;
  const int lane = t & 31;
  const int wav  = t >> 5;
  const int wm   = wav & 1;    // 2 wave rows  -> 64 M each
  const int wn   = wav >> 1;   // 4 wave cols  -> 32 N each

  const int m0 = blockIdx.y * BM;
  const int n0 = blockIdx.x * BN;

  // async staging mapping: 4 passes of 32 rows x 64 cols, 16 B per lane
  const int brow = t >> 3;          // 0..31
  const int bcol = (t & 7) * 8;     // element offset (16 B chunks)
  unsigned ldsA0[4], ldsB0[4], aOff[4], bOff[4];
  #pragma unroll
  for (int p = 0; p < 4; ++p) {
    const int rr = p * 32 + brow;
    ldsA0[p] = (unsigned)(uintptr_t)&sA[0][rr * LDK + bcol];
    ldsB0[p] = (unsigned)(uintptr_t)&sB[0][rr * LDK + bcol];
    aOff[p]  = (unsigned)(((m0 + rr) * IN_DIM + bcol) * 2u);
    bOff[p]  = (unsigned)(((n0 + rr) * IN_DIM + bcol) * 2u);
  }
  const unsigned long long xbBase = (unsigned long long)(uintptr_t)xb;
  const unsigned long long wbBase = (unsigned long long)(uintptr_t)wb;

  auto stage = [&](int buf, unsigned kByte) {
    const unsigned sel = buf ? (unsigned)BUFB : 0u;   // uniform buffer offset
    #pragma unroll
    for (int p = 0; p < 4; ++p)
      asm volatile("global_load_async_to_lds_b128 %0, %1, %2 offset:0"
                   :: "v"(ldsA0[p] + sel), "v"(aOff[p] + kByte), "s"(xbBase)
                   : "memory");
    #pragma unroll
    for (int p = 0; p < 4; ++p)
      asm volatile("global_load_async_to_lds_b128 %0, %1, %2 offset:0"
                   :: "v"(ldsB0[p] + sel), "v"(bOff[p] + kByte), "s"(wbBase)
                   : "memory");
  };

  // ---- WMMA fragment loads (ISA 7.12.2 VGPR layouts)
  const int fr = lane & 15;
  const int hi = lane >> 4;                 // 0 for lanes 0-15, 1 for 16-31
  // A (16x32, bf16): lane m: halves 0-7 = K[0..7]+8*hi, halves 8-15 = +16
  auto loadA = [&](const unsigned short* s, int row, int kb) -> v16bf {
    union { v16bf v; v8bf h[2]; } f;
    const unsigned short* p = s + row * LDK + kb + hi * 8;
    f.h[0] = *(const v8bf*)(p);
    f.h[1] = *(const v8bf*)(p + 16);
    return f.v;
  };
  // B (32x16, bf16): lane n: halves 0-7 = K[0..7]+8*hi, halves 8-15 = +16
  auto loadB = [&](const unsigned short* s, int row, int kb) -> v16bf {
    union { v16bf v; v8bf h[2]; } f;
    const unsigned short* p = s + row * LDK + kb + hi * 8;
    f.h[0] = *(const v8bf*)(p);
    f.h[1] = *(const v8bf*)(p + 16);
    return f.v;
  };

  v8f acc[4][2];
  const v8f vzero = {0.f, 0.f, 0.f, 0.f, 0.f, 0.f, 0.f, 0.f};
  #pragma unroll
  for (int mt = 0; mt < 4; ++mt)
    #pragma unroll
    for (int nt = 0; nt < 2; ++nt) acc[mt][nt] = vzero;

  stage(0, 0u);
  asm volatile("s_wait_asynccnt 0x0" ::: "memory");
  __syncthreads();

  const int NT = IN_DIM / BK;  // 64 K-tiles
  for (int kt = 0; kt < NT; ++kt) {
    const int cur = kt & 1;
    if (kt + 1 < NT)
      stage(cur ^ 1, (unsigned)((kt + 1) * BK * 2));  // engine fills next buf

    #pragma unroll
    for (int ks = 0; ks < 2; ++ks) {            // BK=64 -> two K=32 steps
      const int kb = ks * 32;
      v16bf af[4], bfm[2];
      #pragma unroll
      for (int mt = 0; mt < 4; ++mt)
        af[mt] = loadA(sA[cur], wm * 64 + mt * 16 + fr, kb);
      #pragma unroll
      for (int nt = 0; nt < 2; ++nt)
        bfm[nt] = loadB(sB[cur], wn * 32 + nt * 16 + fr, kb);
      #pragma unroll
      for (int mt = 0; mt < 4; ++mt)
        #pragma unroll
        for (int nt = 0; nt < 2; ++nt)
          acc[mt][nt] = __builtin_amdgcn_wmma_f32_16x16x32_bf16(
              false, af[mt], false, bfm[nt], (short)0, acc[mt][nt],
              false, false);
    }

    if (kt + 1 < NT) {
      asm volatile("s_wait_asynccnt 0x0" ::: "memory");
      __syncthreads();                               // one barrier per K-tile
    }
  }

  // ---- epilogue: C layout = lane(l<16:N=l, M=vgpr | l>=16: M=vgpr+8)
  const int mbase = hi * 8;
  #pragma unroll
  for (int mt = 0; mt < 4; ++mt) {
    #pragma unroll
    for (int nt = 0; nt < 2; ++nt) {
      const int n = n0 + wn * 32 + nt * 16 + fr;
      const float bv = bias[n];
      const size_t base =
          (size_t)(m0 + wm * 64 + mt * 16 + mbase) * OUT_DIM + n;
      #pragma unroll
      for (int r8 = 0; r8 < 8; ++r8)
        out[base + (size_t)r8 * OUT_DIM] = acc[mt][nt][r8] + bv;
    }
  }
}

// ---------------- Phase 2 (fallback, small ws): f32 A converted in-kernel ---
__global__ __launch_bounds__(256) void gemm_wmma_bf16_kernel(
    const float* __restrict__ x, const unsigned short* __restrict__ wb,
    const float* __restrict__ bias, float* __restrict__ out) {
  __shared__ __align__(16) unsigned short sA[2][BM * LDK];
  __shared__ __align__(16) unsigned short sB[2][BN * LDK];

  const int t    = threadIdx.x;
  const int lane = t & 31;
  const int wav  = t >> 5;
  const int wm   = wav & 1;
  const int wn   = wav >> 1;

  const int m0 = blockIdx.y * BM;
  const int n0 = blockIdx.x * BN;

  const int lrow = t >> 4;
  const int lcol = (t & 15) * 4;
  const float* xp = x + (size_t)(m0 + lrow) * IN_DIM + lcol;

  f32x4 aReg[8];
  auto prefetchA = [&](int k0) {
    #pragma unroll
    for (int p = 0; p < 8; ++p)
      aReg[p] = *(const f32x4*)(xp + (size_t)(p * 16) * IN_DIM + k0);
  };
  auto stageA = [&](int buf) {
    #pragma unroll
    for (int p = 0; p < 8; ++p) {
      bf16x4 c;
      c[0] = (__bf16)aReg[p][0]; c[1] = (__bf16)aReg[p][1];
      c[2] = (__bf16)aReg[p][2]; c[3] = (__bf16)aReg[p][3];
      *(bf16x4*)&sA[buf][(lrow + p * 16) * LDK + lcol] = c;
    }
  };

  const int brow = t >> 3;
  const int bcol = (t & 7) * 8;
  unsigned ldsB0[4], bOff[4];
  #pragma unroll
  for (int p = 0; p < 4; ++p) {
    ldsB0[p] = (unsigned)(uintptr_t)&sB[0][(p * 32 + brow) * LDK + bcol];
    bOff[p]  = (unsigned)(((n0 + p * 32 + brow) * IN_DIM + bcol) * 2u);
  }
  const unsigned long long wbBase = (unsigned long long)(uintptr_t)wb;

  auto asyncB = [&](int buf, unsigned kByte) {
    const unsigned sel = buf ? (unsigned)BUFB : 0u;
    #pragma unroll
    for (int p = 0; p < 4; ++p)
      asm volatile("global_load_async_to_lds_b128 %0, %1, %2 offset:0"
                   :: "v"(ldsB0[p] + sel), "v"(bOff[p] + kByte), "s"(wbBase)
                   : "memory");
  };

  const int fr = lane & 15;
  const int hi = lane >> 4;
  auto loadA = [&](const unsigned short* s, int row, int kb) -> v16bf {
    union { v16bf v; v8bf h[2]; } f;
    const unsigned short* p = s + row * LDK + kb + hi * 8;
    f.h[0] = *(const v8bf*)(p);
    f.h[1] = *(const v8bf*)(p + 16);
    return f.v;
  };
  auto loadB = [&](const unsigned short* s, int row, int kb) -> v16bf {
    union { v16bf v; v8bf h[2]; } f;
    const unsigned short* p = s + row * LDK + kb + hi * 8;
    f.h[0] = *(const v8bf*)(p);
    f.h[1] = *(const v8bf*)(p + 16);
    return f.v;
  };

  v8f acc[4][2];
  const v8f vzero = {0.f, 0.f, 0.f, 0.f, 0.f, 0.f, 0.f, 0.f};
  #pragma unroll
  for (int mt = 0; mt < 4; ++mt)
    #pragma unroll
    for (int nt = 0; nt < 2; ++nt) acc[mt][nt] = vzero;

  asyncB(0, 0u);
  prefetchA(0);
  stageA(0);
  asm volatile("s_wait_asynccnt 0x0" ::: "memory");
  __syncthreads();

  const int NT = IN_DIM / BK;
  for (int kt = 0; kt < NT; ++kt) {
    const int cur = kt & 1;
    if (kt + 1 < NT) {
      asyncB(cur ^ 1, (unsigned)((kt + 1) * BK * 2));
      prefetchA((kt + 1) * BK);
    }
    #pragma unroll
    for (int ks = 0; ks < 2; ++ks) {
      const int kb = ks * 32;
      v16bf af[4], bfm[2];
      #pragma unroll
      for (int mt = 0; mt < 4; ++mt)
        af[mt] = loadA(sA[cur], wm * 64 + mt * 16 + fr, kb);
      #pragma unroll
      for (int nt = 0; nt < 2; ++nt)
        bfm[nt] = loadB(sB[cur], wn * 32 + nt * 16 + fr, kb);
      #pragma unroll
      for (int mt = 0; mt < 4; ++mt)
        #pragma unroll
        for (int nt = 0; nt < 2; ++nt)
          acc[mt][nt] = __builtin_amdgcn_wmma_f32_16x16x32_bf16(
              false, af[mt], false, bfm[nt], (short)0, acc[mt][nt],
              false, false);
    }
    if (kt + 1 < NT) {
      stageA(cur ^ 1);
      asm volatile("s_wait_asynccnt 0x0" ::: "memory");
      __syncthreads();
    }
  }

  const int mbase = hi * 8;
  #pragma unroll
  for (int mt = 0; mt < 4; ++mt) {
    #pragma unroll
    for (int nt = 0; nt < 2; ++nt) {
      const int n = n0 + wn * 32 + nt * 16 + fr;
      const float bv = bias[n];
      const size_t base =
          (size_t)(m0 + wm * 64 + mt * 16 + mbase) * OUT_DIM + n;
      #pragma unroll
      for (int r8 = 0; r8 < 8; ++r8)
        out[base + (size_t)r8 * OUT_DIM] = acc[mt][nt][r8] + bv;
    }
  }
}

extern "C" void kernel_launch(void* const* d_in, const int* in_sizes, int n_in,
                              void* d_out, int out_size, void* d_ws,
                              size_t ws_size, hipStream_t stream) {
  (void)in_sizes; (void)n_in; (void)out_size;
  const float* x     = (const float*)d_in[0];
  const float* w     = (const float*)d_in[1];
  const float* bias  = (const float*)d_in[2];
  const int*   basis = (const int*)d_in[3];
  float* out = (float*)d_out;

  const size_t wbBytes = (size_t)OUT_DIM * IN_DIM * 2;   // 32 MiB
  const size_t xbBytes = (size_t)NROWS   * IN_DIM * 2;   // 64 MiB
  unsigned short* wb = (unsigned short*)d_ws;

  hipLaunchKernelGGL(binary_decompose_kernel, dim3(OUT_DIM), dim3(256), 0,
                     stream, w, basis, wb);

  if (ws_size >= wbBytes + xbBytes) {
    unsigned short* xbuf = (unsigned short*)((char*)d_ws + wbBytes);
    hipLaunchKernelGGL(cvt_x_bf16_kernel,
                       dim3((NROWS * (size_t)IN_DIM) / (256 * 8)), dim3(256),
                       0, stream, x, xbuf);
    hipLaunchKernelGGL(gemm_wmma_async_kernel,
                       dim3(OUT_DIM / BN, NROWS / BM), dim3(256), 0, stream,
                       xbuf, wb, bias, out);
  } else {
    hipLaunchKernelGGL(gemm_wmma_bf16_kernel,
                       dim3(OUT_DIM / BN, NROWS / BM), dim3(256), 0, stream,
                       x, wb, bias, out);
  }
}
